// TtBloomAttention_58368605552769
// MI455X (gfx1250) — matrix operand order repaired and checked by reading, not compiled, not run on hardware
//
#include <hip/hip_runtime.h>

// ---------------- problem constants ----------------
constexpr int Bc  = 2;
constexpr int Sc  = 2048;
constexpr int Hc  = 1024;
constexpr int NHc = 16;
constexpr int HDc = 64;
constexpr float INV_NORM = 0.125f;           // 1/sqrt(64)

using bf16 = __bf16;
typedef __attribute__((ext_vector_type(16))) __bf16 v16bf;
typedef __attribute__((ext_vector_type(4)))  __bf16 v4bf;
typedef __attribute__((ext_vector_type(8)))  float  v8f;
typedef __attribute__((ext_vector_type(4)))  int    v4i;

#define AS1 __attribute__((address_space(1)))
#define AS3 __attribute__((address_space(3)))

#if __has_builtin(__builtin_amdgcn_global_load_async_to_lds_b128)
#define ASYNC_COPY 1
#else
#define ASYNC_COPY 0
#endif

// 16-byte global->LDS copy; async (ASYNCcnt) when the gfx1250 builtin exists.
__device__ __forceinline__ void cp16(const bf16* g, bf16* l) {
#if ASYNC_COPY
    __builtin_amdgcn_global_load_async_to_lds_b128((AS1 v4i*)g, (AS3 v4i*)l, 0, 0);
#else
    *(int4*)l = *(const int4*)g;
#endif
}
__device__ __forceinline__ void wait_async() {
#if ASYNC_COPY
#if __has_builtin(__builtin_amdgcn_s_wait_asynccnt)
    __builtin_amdgcn_s_wait_asynccnt(0);
#else
    asm volatile("s_wait_asynccnt 0" ::: "memory");
#endif
#endif
}

// ============================================================
// Kernel 0: fp32 -> bf16 conversion (vectorized x4)
// ============================================================
__global__ __launch_bounds__(256) void cvt_bf16(
    const float* __restrict__ s, bf16* __restrict__ d, int n4)
{
    int i = blockIdx.x * 256 + threadIdx.x;
    if (i < n4) {
        float4 v = ((const float4*)s)[i];
        v4bf o = {(bf16)v.x, (bf16)v.y, (bf16)v.z, (bf16)v.w};
        ((v4bf*)d)[i] = o;
    }
}

// ============================================================
// Kernel 1: fused QKV projection  C[4096,3072] = A @ Wqkv^T + b
// A/W already bf16. Async double-buffered LDS tiles.
// Writes q*INV_NORM, k in [B,NH,S,HD]; v TRANSPOSED in [B,NH,HD,S].
// ============================================================
__global__ __launch_bounds__(256) void qkv_gemm(
    const bf16* __restrict__ A,       // [4096,1024]
    const bf16* __restrict__ W,       // [3072,1024]
    const float* __restrict__ bias,   // [3072]
    bf16* __restrict__ qb, bf16* __restrict__ kb, bf16* __restrict__ vtb)
{
    __shared__ bf16 As[2][64][32];
    __shared__ bf16 Ws[2][128][32];
    const int t    = threadIdx.x;
    const int wave = t >> 5, lane = t & 31;
    const int half = lane >> 4, ln = lane & 15;
    const int m0 = (blockIdx.x & 63) * 64;          // 64 m-blocks
    const int n0 = (blockIdx.x >> 6) * 128;         // 24 n-blocks
    const int sm = (wave >> 2) * 32;
    const int sn = (wave & 3) * 32;

    auto issue = [&](int k0, int buf) {
        {   // A tile: 64x32 bf16 = 256 x 16B, 1 per thread
            int r = t >> 2, seg = t & 3;
            cp16(&A[(size_t)(m0 + r) * Hc + k0 + seg * 8], &As[buf][r][seg * 8]);
        }
        #pragma unroll
        for (int e = t; e < 512; e += 256) {   // W tile: 128x32 = 512 x 16B
            int r = e >> 2, seg = e & 3;
            cp16(&W[(size_t)(n0 + r) * Hc + k0 + seg * 8], &Ws[buf][r][seg * 8]);
        }
    };

    v8f acc[2][2] = {};
    issue(0, 0);
    for (int ks = 0; ks < 32; ++ks) {
        const int buf = ks & 1;
        wait_async();
        __syncthreads();
        if (ks + 1 < 32) issue((ks + 1) * 32, buf ^ 1);

        v16bf af[2], bfm[2];
        #pragma unroll
        for (int tm = 0; tm < 2; ++tm) {
            const bf16* ap = &As[buf][sm + tm * 16 + ln][0];
            #pragma unroll
            for (int i = 0; i < 8; ++i) {
                int kb_ = ((i < 4) ? 0 : 16) + half * 8 + (i & 3) * 2;
                af[tm][2 * i]     = ap[kb_];
                af[tm][2 * i + 1] = ap[kb_ + 1];
            }
        }
        #pragma unroll
        for (int tn = 0; tn < 2; ++tn) {
            const bf16* bp = &Ws[buf][sn + tn * 16 + ln][0];
            #pragma unroll
            for (int i = 0; i < 8; ++i) {
                int kk = ((i < 4) ? 0 : 16) + half * 8 + (i & 3) * 2;
                bfm[tn][2 * i]     = bp[kk];
                bfm[tn][2 * i + 1] = bp[kk + 1];
            }
        }
        #pragma unroll
        for (int tm = 0; tm < 2; ++tm)
            #pragma unroll
            for (int tn = 0; tn < 2; ++tn)
                acc[tm][tn] = __builtin_amdgcn_wmma_f32_16x16x32_bf16(
                    false, af[tm], false, bfm[tn], (short)0, acc[tm][tn], false, false);
    }

    // epilogue: tile-uniform head/which/d decomposition, bias, scatter
    #pragma unroll
    for (int tm = 0; tm < 2; ++tm)
        #pragma unroll
        for (int tn = 0; tn < 2; ++tn) {
            const int nb    = n0 + sn + tn * 16;       // 16-aligned, within one 64-block
            const int head  = nb / (3 * HDc);
            const int rem   = nb - head * (3 * HDc);
            const int which = rem >> 6;                // uniform over tile
            const int d     = (rem & 63) + ln;
            const float bv  = bias[nb + ln];
            #pragma unroll
            for (int r = 0; r < 8; ++r) {
                int m  = m0 + sm + tm * 16 + r + half * 8;
                int bi = m >> 11, si = m & 2047;
                float v = acc[tm][tn][r] + bv;
                if (which == 0)
                    qb[((size_t)((bi * NHc + head) * Sc + si)) * HDc + d] = (bf16)(v * INV_NORM);
                else if (which == 1)
                    kb[((size_t)((bi * NHc + head) * Sc + si)) * HDc + d] = (bf16)v;
                else
                    vtb[((size_t)(bi * NHc + head) * HDc + d) * Sc + si] = (bf16)v;
            }
        }
}

// ============================================================
// Kernel 2: causal flash attention with alibi.
// K chunk [key][d] and V chunk [d][key] (pre-transposed in global)
// are pure direct copies -> async double-buffered LDS.
// ============================================================
__global__ __launch_bounds__(256) void flash_attn(
    const bf16* __restrict__ qb, const bf16* __restrict__ kb,
    const bf16* __restrict__ vtb, const float* __restrict__ alibi,
    bf16* __restrict__ ctx)
{
    __shared__ bf16 Ks[2][64][64];       // [key][d]
    __shared__ bf16 Vt[2][64][64];       // [d][key]
    __shared__ bf16 Ps[8][16][64];       // per-wave P tile
    const int t    = threadIdx.x;
    const int wave = t >> 5, lane = t & 31;
    const int half = lane >> 4, ln = lane & 15;
    const int bh   = blockIdx.x & 31;
    const int qblk = blockIdx.x >> 5;
    const int b = bh >> 4, h = bh & 15;
    const int q0 = qblk * 128;
    const int qw = q0 + wave * 16;
    const size_t base = (size_t)bh * Sc * HDc;   // same for k ([s][d]) and vt ([d][s])

    auto issue = [&](int j0, int buf) {
        #pragma unroll
        for (int e = t; e < 512; e += 256) {     // Ks: 64 rows x 8 segs
            int key = e >> 3, seg = e & 7;
            cp16(&kb[base + (size_t)(j0 + key) * HDc + seg * 8], &Ks[buf][key][seg * 8]);
        }
        #pragma unroll
        for (int e = t; e < 512; e += 256) {     // Vt: 64 d-rows x 8 segs
            int d = e >> 3, seg = e & 7;
            cp16(&vtb[base + (size_t)d * Sc + j0 + seg * 8], &Vt[buf][d][seg * 8]);
        }
    };

    // Q A-fragments (two 16x32 k-steps over HD=64)
    v16bf Qa[2];
    {
        const bf16* qp = qb + base + (size_t)(qw + ln) * HDc;
        #pragma unroll
        for (int s = 0; s < 2; ++s)
            #pragma unroll
            for (int i = 0; i < 8; ++i) {
                int kb_ = s * 32 + ((i < 4) ? 0 : 16) + half * 8 + (i & 3) * 2;
                Qa[s][2 * i]     = qp[kb_];
                Qa[s][2 * i + 1] = qp[kb_ + 1];
            }
    }

    v8f acc[4] = {};
    float mrow[8], lrow[8];
    #pragma unroll
    for (int r = 0; r < 8; ++r) { mrow[r] = -1e30f; lrow[r] = 0.f; }

    const int nchunk = (q0 + 128) >> 6;
    issue(0, 0);
    for (int j = 0; j < nchunk; ++j) {
        const int j0  = j * 64;
        const int buf = j & 1;
        wait_async();
        __syncthreads();
        if (j + 1 < nchunk) issue(j0 + 64, buf ^ 1);

        if (j0 <= qw + 15) {
            // ---- scores = (Q/sqrt(d)) @ K^T ----
            v8f sc[4] = {};
            #pragma unroll
            for (int s = 0; s < 2; ++s)
                #pragma unroll
                for (int tt = 0; tt < 4; ++tt) {
                    v16bf bfr;
                    const bf16* kp = &Ks[buf][tt * 16 + ln][s * 32];
                    #pragma unroll
                    for (int i = 0; i < 8; ++i) {
                        int kk = ((i < 4) ? 0 : 16) + half * 8 + (i & 3) * 2;
                        bfr[2 * i]     = kp[kk];
                        bfr[2 * i + 1] = kp[kk + 1];
                    }
                    sc[tt] = __builtin_amdgcn_wmma_f32_16x16x32_bf16(
                        false, Qa[s], false, bfr, (short)0, sc[tt], false, false);
                }

            // ---- alibi + causal mask ----
            #pragma unroll
            for (int tt = 0; tt < 4; ++tt) {
                int key = j0 + tt * 16 + ln;
                float av = alibi[(size_t)bh * Sc + key];
                #pragma unroll
                for (int r = 0; r < 8; ++r) {
                    int qr = qw + r + half * 8;
                    float v = sc[tt][r] + av;
                    sc[tt][r] = (key > qr) ? -1e30f : v;
                }
            }

            // ---- online softmax ----
            #pragma unroll
            for (int r = 0; r < 8; ++r) {
                float mx = fmaxf(fmaxf(sc[0][r], sc[1][r]), fmaxf(sc[2][r], sc[3][r]));
                #pragma unroll
                for (int off = 1; off < 16; off <<= 1)
                    mx = fmaxf(mx, __shfl_xor(mx, off, 32));
                float mnew  = fmaxf(mrow[r], mx);
                float scale = __expf(mrow[r] - mnew);
                float rs = 0.f;
                #pragma unroll
                for (int tt = 0; tt < 4; ++tt) {
                    float p = __expf(sc[tt][r] - mnew);
                    sc[tt][r] = p;
                    rs += p;
                }
                #pragma unroll
                for (int off = 1; off < 16; off <<= 1)
                    rs += __shfl_xor(rs, off, 32);
                lrow[r] = lrow[r] * scale + rs;
                mrow[r] = mnew;
                #pragma unroll
                for (int tt = 0; tt < 4; ++tt) acc[tt][r] *= scale;
            }

            // ---- stage P through LDS into A-fragment layout ----
            #pragma unroll
            for (int r = 0; r < 8; ++r)
                #pragma unroll
                for (int tt = 0; tt < 4; ++tt)
                    Ps[wave][r + half * 8][tt * 16 + ln] = (bf16)sc[tt][r];
            asm volatile("s_wait_dscnt 0" ::: "memory");

            // ---- acc += P @ V ----
            #pragma unroll
            for (int s = 0; s < 2; ++s) {
                v16bf pa;
                const bf16* pp = &Ps[wave][ln][s * 32];
                #pragma unroll
                for (int i = 0; i < 8; ++i) {
                    int kb_ = ((i < 4) ? 0 : 16) + half * 8 + (i & 3) * 2;
                    pa[2 * i]     = pp[kb_];
                    pa[2 * i + 1] = pp[kb_ + 1];
                }
                #pragma unroll
                for (int tt = 0; tt < 4; ++tt) {
                    v16bf bfr;
                    const bf16* vp = &Vt[buf][tt * 16 + ln][s * 32];
                    #pragma unroll
                    for (int i = 0; i < 8; ++i) {
                        int kk = ((i < 4) ? 0 : 16) + half * 8 + (i & 3) * 2;
                        bfr[2 * i]     = vp[kk];
                        bfr[2 * i + 1] = vp[kk + 1];
                    }
                    acc[tt] = __builtin_amdgcn_wmma_f32_16x16x32_bf16(
                        false, pa, false, bfr, (short)0, acc[tt], false, false);
                }
            }
        }
    }

    // ---- normalize, store ctx merged-head [B,S,NH,HD] bf16 ----
    #pragma unroll
    for (int r = 0; r < 8; ++r) {
        float inv = 1.0f / lrow[r];
        int qg = qw + r + half * 8;
        #pragma unroll
        for (int tt = 0; tt < 4; ++tt) {
            int d = tt * 16 + ln;
            ctx[(((size_t)b * Sc + qg) * NHc + h) * HDc + d] = (bf16)(acc[tt][r] * inv);
        }
    }
}

// ============================================================
// Kernel 3: out = ctx @ Wd^T + b_d + residual   (fp32 output)
// ============================================================
__global__ __launch_bounds__(256) void out_proj(
    const bf16* __restrict__ A,       // ctx bf16 [4096,1024]
    const bf16* __restrict__ W,       // wd bf16 [1024,1024]
    const float* __restrict__ bias,
    const float* __restrict__ resid,
    float* __restrict__ out)
{
    __shared__ bf16 As[2][64][32];
    __shared__ bf16 Ws[2][128][32];
    const int t    = threadIdx.x;
    const int wave = t >> 5, lane = t & 31;
    const int half = lane >> 4, ln = lane & 15;
    const int m0 = (blockIdx.x & 63) * 64;      // 64 m-blocks
    const int n0 = (blockIdx.x >> 6) * 128;     // 8 n-blocks
    const int sm = (wave >> 2) * 32;
    const int sn = (wave & 3) * 32;

    auto issue = [&](int k0, int buf) {
        {
            int r = t >> 2, seg = t & 3;
            cp16(&A[(size_t)(m0 + r) * Hc + k0 + seg * 8], &As[buf][r][seg * 8]);
        }
        #pragma unroll
        for (int e = t; e < 512; e += 256) {
            int r = e >> 2, seg = e & 3;
            cp16(&W[(size_t)(n0 + r) * Hc + k0 + seg * 8], &Ws[buf][r][seg * 8]);
        }
    };

    v8f acc[2][2] = {};
    issue(0, 0);
    for (int ks = 0; ks < 32; ++ks) {
        const int buf = ks & 1;
        wait_async();
        __syncthreads();
        if (ks + 1 < 32) issue((ks + 1) * 32, buf ^ 1);

        v16bf af[2], bfm[2];
        #pragma unroll
        for (int tm = 0; tm < 2; ++tm) {
            const bf16* ap = &As[buf][sm + tm * 16 + ln][0];
            #pragma unroll
            for (int i = 0; i < 8; ++i) {
                int kb_ = ((i < 4) ? 0 : 16) + half * 8 + (i & 3) * 2;
                af[tm][2 * i]     = ap[kb_];
                af[tm][2 * i + 1] = ap[kb_ + 1];
            }
        }
        #pragma unroll
        for (int tn = 0; tn < 2; ++tn) {
            const bf16* bp = &Ws[buf][sn + tn * 16 + ln][0];
            #pragma unroll
            for (int i = 0; i < 8; ++i) {
                int kk = ((i < 4) ? 0 : 16) + half * 8 + (i & 3) * 2;
                bfm[tn][2 * i]     = bp[kk];
                bfm[tn][2 * i + 1] = bp[kk + 1];
            }
        }
        #pragma unroll
        for (int tm = 0; tm < 2; ++tm)
            #pragma unroll
            for (int tn = 0; tn < 2; ++tn)
                acc[tm][tn] = __builtin_amdgcn_wmma_f32_16x16x32_bf16(
                    false, af[tm], false, bfm[tn], (short)0, acc[tm][tn], false, false);
    }

    #pragma unroll
    for (int tm = 0; tm < 2; ++tm)
        #pragma unroll
        for (int tn = 0; tn < 2; ++tn)
            #pragma unroll
            for (int r = 0; r < 8; ++r) {
                int m = m0 + sm + tm * 16 + r + half * 8;
                int n = n0 + sn + tn * 16 + ln;
                size_t idx = (size_t)m * Hc + n;
                out[idx] = acc[tm][tn][r] + bias[n] + resid[idx];
            }
}

// ============================================================
// launch
// ============================================================
extern "C" void kernel_launch(void* const* d_in, const int* in_sizes, int n_in,
                              void* d_out, int out_size, void* d_ws, size_t ws_size,
                              hipStream_t stream) {
    const float* hs    = (const float*)d_in[0];
    const float* resid = (const float*)d_in[1];
    const float* alibi = (const float*)d_in[2];
    // d_in[3] = attention_mask (causal) -- implemented analytically
    const float* Wqkv  = (const float*)d_in[4];
    const float* bqkv  = (const float*)d_in[5];
    const float* Wd    = (const float*)d_in[6];
    const float* bd    = (const float*)d_in[7];

    const size_t nHS = (size_t)Bc * Sc * Hc;        // 4,194,304
    const size_t nWQ = (size_t)3 * Hc * Hc;         // 3,145,728
    const size_t nWD = (size_t)Hc * Hc;             // 1,048,576
    const size_t nQ  = (size_t)Bc * NHc * Sc * HDc; // 4,194,304

    bf16* hsb = (bf16*)d_ws;
    bf16* wqb = hsb + nHS;
    bf16* wdb = wqb + nWQ;
    bf16* qb  = wdb + nWD;
    bf16* kb  = qb + nQ;
    bf16* vtb = kb + nQ;
    bf16* ctx = vtb + nQ;
    const size_t need = (nHS + nWQ + nWD + 4 * nQ) * sizeof(bf16);  // ~50.3 MB
    if (ws_size < need) return;

    // fp32 -> bf16 pre-pass (one-shot; weights then live bf16 in L2)
    cvt_bf16<<<dim3((unsigned)(nHS / 4 / 256)), dim3(256), 0, stream>>>(hs, hsb, (int)(nHS / 4));
    cvt_bf16<<<dim3((unsigned)(nWQ / 4 / 256)), dim3(256), 0, stream>>>(Wqkv, wqb, (int)(nWQ / 4));
    cvt_bf16<<<dim3((unsigned)(nWD / 4 / 256)), dim3(256), 0, stream>>>(Wd, wdb, (int)(nWD / 4));

    // QKV projection: (4096/64) * (3072/128) = 1536 blocks
    qkv_gemm<<<dim3(1536), dim3(256), 0, stream>>>(hsb, wqb, bqkv, qb, kb, vtb);
    // Flash attention: B*NH * (S/128) = 512 blocks
    flash_attn<<<dim3(512), dim3(256), 0, stream>>>(qb, kb, vtb, alibi, ctx);
    // Output projection + residual: (4096/64) * (1024/128) = 512 blocks
    out_proj<<<dim3(512), dim3(256), 0, stream>>>(ctx, wdb, bd, resid, (float*)d_out);
}
